// ResOffsetEstimator_52424370815575
// MI455X (gfx1250) — matrix operand order repaired and checked by reading, not compiled, not run on hardware
//
#include <hip/hip_runtime.h>
#include <hip/hip_bf16.h>
#include <math.h>

typedef __attribute__((ext_vector_type(16))) _Float16 v16h;
typedef __attribute__((ext_vector_type(8)))  _Float16 v8h;
typedef __attribute__((ext_vector_type(2)))  _Float16 v2h;
typedef __attribute__((ext_vector_type(8)))  float    v8f;
typedef _Float16 h16;

constexpr int   NBATCH = 2;
constexpr int   HH = 64, WW = 64;
constexpr int   HWC = HH * WW;              // 4096 (multiple of 16)
constexpr float BNF_C = 0.9999950000374997f; // 1/sqrt(1+1e-5)

__device__ __forceinline__ float sigm(float x) { return 1.f / (1.f + __expf(-x)); }
__device__ __forceinline__ int   clampi(int v, int lo, int hi) { return v < lo ? lo : (v > hi ? hi : v); }

// f16 activation buffers use K-pair interleaved layout:
//   element (b, c, pix) of a [B, Ct, HW] tensor (Ct even) lives at
//   ((b*Ct/2 + c/2)*HW + pix)*2 + (c&1)
// so channel pair (2c, 2c+1) at one pixel is a single aligned dword.
__device__ __forceinline__ size_t hidx(int b, int Ct, int c, int pix) {
    return ((((size_t)b * (Ct >> 1) + (c >> 1)) * HWC + pix) << 1) + (c & 1);
}

// tap index -> resample slot j (RESAMPLE_IDX = 12,0,2,4,10,14,20,22,24)
__device__ __constant__ int RINV[25] = { 1,-1, 2,-1, 3, -1,-1,-1,-1,-1,
                                         4,-1, 0,-1, 5, -1,-1,-1,-1,-1,
                                         6,-1, 7,-1, 8 };

// ---------------------------------------------------------------------------
// WMMA GEMM on zero-padded operands (no bounds checks in the K loop).
//   W: [Op, Cp] f16, zero-padded (Op mult 16, Cp mult 32), row-major
//   X: [B, XCt, HW] f16 pair-interleaved, channels XCo..XCo+Cp-1 valid
// A frag = 2 x b128 per K-step; B frag = 8 x b32 per K-step.
// 4 waves / block, each wave owns a 16(O)x16(N) tile (same N, different O).
// ---------------------------------------------------------------------------
__global__ void k_gemm(const h16* __restrict__ Wm,
                       const h16* __restrict__ Xm, int XCt, int XCo,
                       float* __restrict__ Yf, int YfCt, int YfCo,
                       h16*   __restrict__ Yh, int YhCt, int YhCo,
                       const float* __restrict__ addp, int addCt, int addCo,
                       const float* __restrict__ bias,
                       int O, int Cp, int relu, float alpha)
{
    const int lane = threadIdx.x & 31;
    const int wave = threadIdx.x >> 5;          // 0..3
    const int lo = lane & 15, hi = lane >> 4;
    const int n0 = blockIdx.x << 4;
    const int o0 = (blockIdx.y * 4 + wave) << 4;
    if (o0 >= O) return;                        // uniform per wave
    const int b  = blockIdx.z;
    // pair-layout base: batch b, channel offset XCo (even)
    const h16* __restrict__ Xb = Xm + ((size_t)b * (XCt >> 1) + (XCo >> 1)) * HWC * 2;
    const int col = n0 + lo;
    const h16* __restrict__ Wr = Wm + (size_t)(o0 + lo) * Cp + (hi << 3);

    v8f acc = {};
    for (int k = 0; k < Cp; k += 32) {
        const v8h a0 = *(const v8h*)(Wr + k);        // K = hi*8 + 0..7
        const v8h a1 = *(const v8h*)(Wr + k + 16);   // K = 16 + hi*8 + 0..7
        v16h a, bf;
        #pragma unroll
        for (int t = 0; t < 8; ++t) { a[t] = a0[t]; a[8 + t] = a1[t]; }
        #pragma unroll
        for (int v = 0; v < 8; ++v) {
            const int kp = ((k + (hi << 3)) >> 1) + v + ((v >> 2) << 2);   // channel-pair index
            const v2h p = *(const v2h*)(Xb + ((size_t)kp * HWC + col) * 2);
            bf[2 * v] = p[0]; bf[2 * v + 1] = p[1];
        }
        acc = __builtin_amdgcn_wmma_f32_16x16x32_f16(false, a, false, bf,
                                                     (short)0, acc, false, false);
    }
    #pragma unroll
    for (int v = 0; v < 8; ++v) {
        const int row = o0 + v + (hi << 3);
        if (row >= O) continue;
        float y = acc[v] * alpha;
        if (bias) y += bias[row];
        if (addp) y += addp[((size_t)b * addCt + addCo + row) * HWC + col];
        if (relu) y = fmaxf(y, 0.f);
        if (Yf) Yf[((size_t)b * YfCt + YfCo + row) * HWC + col] = y;
        if (Yh) Yh[hidx(b, YhCt, YhCo + row, col)] = (h16)y;
    }
}

// ---------------------------------------------------------------------------
// helpers
// ---------------------------------------------------------------------------
// weight f32 [O,C] -> padded f16 [Op,Cp] (zero fill), row-major
__global__ void k_cvtW(const float* __restrict__ s, h16* __restrict__ d,
                       int O, int C, int Op, int Cp) {
    int i = blockIdx.x * blockDim.x + threadIdx.x;
    if (i >= Op * Cp) return;
    int c = i % Cp, o = i / Cp;
    d[i] = (o < O && c < C) ? (h16)s[(size_t)o * C + c] : (h16)0.f;
}

// zero pad channels [Co, Co+Cz) of pair-layout f16 [B, Ct, HW]
__global__ void k_zero_h(h16* __restrict__ d, int Ct, int Co, int Cz) {
    int i = blockIdx.x * blockDim.x + threadIdx.x;
    int tot = NBATCH * Cz * HWC;
    if (i >= tot) return;
    int pix = i % HWC; int c = (i / HWC) % Cz; int b = i / (HWC * Cz);
    d[hidx(b, Ct, Co + c, pix)] = (h16)0.f;
}

// place f32 [B, sCt, HW] (channels sCo..) into pair-layout f16 [B, dCt, HW] at dCo
__global__ void k_place_h(const float* __restrict__ s, int sCt, int sCo,
                          h16* __restrict__ d, int dCt, int dCo, int Cs) {
    int i = blockIdx.x * blockDim.x + threadIdx.x;
    int tot = NBATCH * Cs * HWC;
    if (i >= tot) return;
    int pix = i % HWC; int c = (i / HWC) % Cs; int b = i / (HWC * Cs);
    float v = s[((size_t)b * sCt + sCo + c) * HWC + pix];
    d[hidx(b, dCt, dCo + c, pix)] = (h16)v;
}

__global__ void k_copy_f(const float* __restrict__ s, float* __restrict__ d, int n) {
    int i = blockIdx.x * blockDim.x + threadIdx.x;
    if (i < n) d[i] = s[i];
}

// overlap = tanh(certainty[:,1]/10) -> f32 buffer + f16 into ffcat ch 320 (Ct=352)
__global__ void k_overlap(const float* __restrict__ cert, float* __restrict__ ov,
                          h16* __restrict__ ffcat) {
    int i = blockIdx.x * blockDim.x + threadIdx.x;
    if (i >= NBATCH * HWC) return;
    int pix = i % HWC, b = i / HWC;
    float v = tanhf(cert[((size_t)b * 2 + 1) * HWC + pix] * 0.1f);
    ov[(size_t)b * HWC + pix] = v;
    ffcat[hidx(b, 352, 320, pix)] = (h16)v;
}

// GLU combine: y = a*sigmoid(b)*BNF ; mode1: leaky(0.05); mode2: sigmoid(y)*3+0.5
__global__ void k_glu(const float* __restrict__ A, const float* __restrict__ Bm,
                      float* __restrict__ of, int oCt, int oCo,
                      h16* __restrict__ oh, int hCt, int hCo,
                      int C, int mode) {
    int i = blockIdx.x * blockDim.x + threadIdx.x;
    int tot = NBATCH * C * HWC;
    if (i >= tot) return;
    int pix = i % HWC; int c = (i / HWC) % C; int b = i / (HWC * C);
    float v = A[i] * sigm(Bm[i]) * BNF_C;
    if (mode == 1) v = v > 0.f ? v : 0.05f * v;
    if (mode == 2) v = sigm(v) * 3.f + 0.5f;
    if (of) of[((size_t)b * oCt + oCo + c) * HWC + pix] = v;
    if (oh) oh[hidx(b, hCt, hCo + c, pix)] = (h16)v;
}

// grid_sample(f2c[B,160,HW], grid=flow) -> f16 into ffcat at ch 160 (Ct=352)
__global__ void k_gsample_flow(const float* __restrict__ F, const float* __restrict__ flow,
                               h16* __restrict__ ffcat) {
    int i = blockIdx.x * blockDim.x + threadIdx.x;
    int tot = NBATCH * 160 * HWC;
    if (i >= tot) return;
    int pix = i % HWC; int c = (i / HWC) % 160; int b = i / (HWC * 160);
    float gx = flow[((size_t)b * 2 + 0) * HWC + pix];
    float gy = flow[((size_t)b * 2 + 1) * HWC + pix];
    float px = (gx + 1.f) * (WW * 0.5f) - 0.5f;
    float py = (gy + 1.f) * (HH * 0.5f) - 0.5f;
    float x0f = floorf(px), y0f = floorf(py);
    int x0 = (int)x0f, y0 = (int)y0f;
    float wx = px - x0f, wy = py - y0f;
    float w00 = (1.f - wx) * (1.f - wy), w10 = wx * (1.f - wy);
    float w01 = (1.f - wx) * wy,        w11 = wx * wy;
    if (!(x0 >= 0 && x0 < WW && y0 >= 0 && y0 < HH)) w00 = 0.f;
    if (!(x0 + 1 >= 0 && x0 + 1 < WW && y0 >= 0 && y0 < HH)) w10 = 0.f;
    if (!(x0 >= 0 && x0 < WW && y0 + 1 >= 0 && y0 + 1 < HH)) w01 = 0.f;
    if (!(x0 + 1 >= 0 && x0 + 1 < WW && y0 + 1 >= 0 && y0 + 1 < HH)) w11 = 0.f;
    int cx0 = clampi(x0, 0, WW - 1), cx1 = clampi(x0 + 1, 0, WW - 1);
    int cy0 = clampi(y0, 0, HH - 1), cy1 = clampi(y0 + 1, 0, HH - 1);
    const float* Fc = F + ((size_t)b * 160 + c) * HWC;
    float v = w00 * Fc[cy0 * WW + cx0] + w10 * Fc[cy0 * WW + cx1]
            + w01 * Fc[cy1 * WW + cx0] + w11 * Fc[cy1 * WW + cx1];
    ffcat[hidx(b, 352, 160 + c, pix)] = (h16)v;
}

// ctxc = [context_new ch192..255 (64) | cert_diff(9) | overlap(1)]  (74 ch, f32)
__global__ void k_build_ctxc(const float* __restrict__ ctxn, const float* __restrict__ ov,
                             float* __restrict__ ctxc) {
    int i = blockIdx.x * blockDim.x + threadIdx.x;
    int tot = NBATCH * 74 * HWC;
    if (i >= tot) return;
    int pix = i % HWC; int c = (i / HWC) % 74; int b = i / (HWC * 74);
    float v;
    if (c < 64) v = ctxn[((size_t)b * 256 + 192 + c) * HWC + pix];
    else if (c < 73) {
        int t = c - 64, dy = t / 3 - 1, dx = t % 3 - 1;
        int y = pix / WW + dy, x = pix % WW + dx;
        v = (x >= 0 && x < WW && y >= 0 && y < HH) ? ov[(size_t)b * HWC + y * WW + x] : 0.f;
    } else v = ov[(size_t)b * HWC + pix];
    ctxc[((size_t)b * 74 + c) * HWC + pix] = v;
}

// dense 3x3 conv: in [B,Ci,HW] f32, w [Co,Ci,3,3] f32, out [B,Co,HW]*alpha
__global__ void k_conv3x3(const float* __restrict__ X, const float* __restrict__ Wt,
                          float* __restrict__ Y, int Ci, int Co, float alpha) {
    int i = blockIdx.x * blockDim.x + threadIdx.x;
    int tot = NBATCH * Co * HWC;
    if (i >= tot) return;
    int pix = i % HWC; int o = (i / HWC) % Co; int b = i / (HWC * Co);
    int y0 = pix / WW, x0 = pix % WW;
    float acc = 0.f;
    for (int c = 0; c < Ci; ++c) {
        const float* Xc = X + ((size_t)b * Ci + c) * HWC;
        const float* wc = Wt + ((size_t)o * Ci + c) * 9;
        #pragma unroll
        for (int t = 0; t < 9; ++t) {
            int y = y0 + t / 3 - 1, x = x0 + t % 3 - 1;
            if (x >= 0 && x < WW && y >= 0 && y < HH)
                acc += wc[t] * Xc[y * WW + x];
        }
    }
    Y[((size_t)b * Co + o) * HWC + pix] = acc * alpha;
}

// softmax over 9 taps (per group g) * |flow neighbor diff|  -> f16 into Xs ch 74..91
__global__ void k_score_sm_sc(const float* __restrict__ score, const float* __restrict__ flow,
                              h16* __restrict__ Xs) {
    int i = blockIdx.x * blockDim.x + threadIdx.x;
    int tot = NBATCH * 2 * HWC;
    if (i >= tot) return;
    int pix = i % HWC; int g = (i / HWC) % 2; int b = i / (HWC * 2);
    float s[9], mx = -1e30f;
    for (int k = 0; k < 9; ++k) {
        s[k] = score[((size_t)b * 18 + g * 9 + k) * HWC + pix];
        mx = fmaxf(mx, s[k]);
    }
    float den = 0.f;
    for (int k = 0; k < 9; ++k) { s[k] = __expf(s[k] - mx); den += s[k]; }
    float fc = flow[((size_t)b * 2 + g) * HWC + pix];
    int y0 = pix / WW, x0 = pix % WW;
    for (int k = 0; k < 9; ++k) {
        int y = y0 + k / 3 - 1, x = x0 + k % 3 - 1;
        float fn = (x >= 0 && x < WW && y >= 0 && y < HH)
                 ? flow[((size_t)b * 2 + g) * HWC + y * WW + x] : 0.f;
        float v = (s[k] / den) * fabsf(fn - fc);
        Xs[hidx(b, 96, 74 + g * 9 + k, pix)] = (h16)v;
    }
}

// ratio channels 92..95 = {1,1,2,2}
__global__ void k_ratio(h16* __restrict__ Xs) {
    int i = blockIdx.x * blockDim.x + threadIdx.x;
    int tot = NBATCH * 4 * HWC;
    if (i >= tot) return;
    int pix = i % HWC; int c = (i / HWC) % 4; int b = i / (HWC * 4);
    float v = (c < 2) ? 1.f : 2.f;
    Xs[hidx(b, 96, 92 + c, pix)] = (h16)v;
}

// depthwise 3x3 (pair-layout f16 in/out, f32 weights [C,9])
__global__ void k_dwconv(const h16* __restrict__ X, const float* __restrict__ Wt,
                         h16* __restrict__ Y, int C, int Cti, int Cto) {
    int i = blockIdx.x * blockDim.x + threadIdx.x;
    int tot = NBATCH * C * HWC;
    if (i >= tot) return;
    int pix = i % HWC; int c = (i / HWC) % C; int b = i / (HWC * C);
    int y0 = pix / WW, x0 = pix % WW;
    const float* wc = Wt + (size_t)c * 9;
    float acc = 0.f;
    #pragma unroll
    for (int t = 0; t < 9; ++t) {
        int y = y0 + t / 3 - 1, x = x0 + t % 3 - 1;
        if (x >= 0 && x < WW && y >= 0 && y < HH)
            acc += wc[t] * (float)X[hidx(b, Cti, c, y * WW + x)];
    }
    Y[hidx(b, Cto, c, pix)] = (h16)acc;
}

// local correlation pass 1
__global__ void k_corr1(const float* __restrict__ f1, const float* __restrict__ f0,
                        const float* __restrict__ flow, const float* __restrict__ scale,
                        float* __restrict__ cv, h16* __restrict__ rs) {
    int i = blockIdx.x * blockDim.x + threadIdx.x;
    int tot = NBATCH * HWC * 25;
    if (i >= tot) return;
    int t = i % 25; int pix = (i / 25) % HWC; int b = i / (25 * HWC);
    float fx = flow[((size_t)b * 2 + 0) * HWC + pix];
    float fy = flow[((size_t)b * 2 + 1) * HWC + pix];
    float sx = scale[((size_t)b * 2 + 0) * HWC + pix] * (4.f / (float)WW);
    float sy = scale[((size_t)b * 2 + 1) * HWC + pix] * (4.f / (float)HH);
    float lwx = -1.f + 0.5f * (float)(t % 5);
    float lwy = -1.f + 0.5f * (float)(t / 5);
    float gx = fx + lwx * sx, gy = fy + lwy * sy;
    bool oob = (gx < -1.f) || (gx > 1.f) || (gy < -1.f) || (gy > 1.f);
    float px = (gx + 1.f) * (WW * 0.5f) - 0.5f;
    float py = (gy + 1.f) * (HH * 0.5f) - 0.5f;
    float x0f = floorf(px), y0f = floorf(py);
    int x0 = (int)x0f, y0 = (int)y0f;
    float wx = px - x0f, wy = py - y0f;
    float w00 = (1.f - wx) * (1.f - wy), w10 = wx * (1.f - wy);
    float w01 = (1.f - wx) * wy,        w11 = wx * wy;
    if (!(x0 >= 0 && x0 < WW && y0 >= 0 && y0 < HH)) w00 = 0.f;
    if (!(x0 + 1 < WW && x0 + 1 >= 0 && y0 >= 0 && y0 < HH)) w10 = 0.f;
    if (!(x0 >= 0 && x0 < WW && y0 + 1 >= 0 && y0 + 1 < HH)) w01 = 0.f;
    if (!(x0 + 1 < WW && x0 + 1 >= 0 && y0 + 1 < HH && y0 + 1 >= 0)) w11 = 0.f;
    int cx0 = clampi(x0, 0, WW - 1), cx1 = clampi(x0 + 1, 0, WW - 1);
    int cy0 = clampi(y0, 0, HH - 1), cy1 = clampi(y0 + 1, 0, HH - 1);
    int i00 = cy0 * WW + cx0, i10 = cy0 * WW + cx1;
    int i01 = cy1 * WW + cx0, i11 = cy1 * WW + cx1;
    const int j = RINV[t];
    const float* F1 = f1 + (size_t)b * 131 * HWC;
    const float* F0 = f0 + (size_t)b * 128 * HWC;
    float acc = 0.f;
    for (int c = 0; c < 131; ++c) {
        const float* Fc = F1 + (size_t)c * HWC;
        float v = w00 * Fc[i00] + w10 * Fc[i10] + w01 * Fc[i01] + w11 * Fc[i11];
        if (j >= 0) rs[(((size_t)b * 9 + j) * 131 + c) * HWC + pix] = (h16)v;
        if (c >= 3) acc += v * F0[(size_t)(c - 3) * HWC + pix];
    }
    float r = oob ? -1.f : tanhf(acc * (1.f / 1280.f));
    cv[((size_t)b * 25 + t) * HWC + pix] = r * 10.f;
}

// correlation pass 2: SV [B,36,HW] from rs
__global__ void k_corr2(const h16* __restrict__ rs, float* __restrict__ SV) {
    int i = blockIdx.x * blockDim.x + threadIdx.x;
    int tot = NBATCH * 9 * HWC;
    if (i >= tot) return;
    int pix = i % HWC; int j = (i / HWC) % 9; int b = i / (HWC * 9);
    const h16* r0 = rs + ((size_t)b * 9 + 0) * 131 * HWC;
    const h16* rj = rs + ((size_t)b * 9 + j) * 131 * HWC;
    float acc = 0.f;
    for (int c = 0; c < 131; ++c)
        acc += (float)r0[(size_t)c * HWC + pix] * (float)rj[(size_t)c * HWC + pix];
    SV[((size_t)b * 36 + 27 + j) * HWC + pix] = tanhf(acc * (1.f / 1280.f));
    for (int c = 0; c < 3; ++c)
        SV[((size_t)b * 36 + c * 9 + j) * HWC + pix] = (float)rj[(size_t)c * HWC + pix];
}

__global__ void k_softplus(const float* __restrict__ s, float* __restrict__ d, int n) {
    int i = blockIdx.x * blockDim.x + threadIdx.x;
    if (i < n) { float x = s[i]; d[i] = x > 20.f ? x : log1pf(__expf(x)); }
}

// skip einsum + softmax over 25 + disp reduction
__global__ void k_decoder_final(const float* __restrict__ m, const float* __restrict__ cv,
                                const float* __restrict__ sp, const float* __restrict__ off,
                                float* __restrict__ disp) {
    int i = blockIdx.x * blockDim.x + threadIdx.x;
    int tot = NBATCH * 2 * HWC;
    if (i >= tot) return;
    int pix = i % HWC; int g = (i / HWC) % 2; int b = i / (HWC * 2);
    float cvv[25];
    for (int c = 0; c < 25; ++c) cvv[c] = cv[((size_t)b * 25 + c) * HWC + pix];
    float vals[25], mx = -1e30f;
    for (int cc = 0; cc < 25; ++cc) {
        float s = 0.f;
        for (int c = 0; c < 25; ++c) s += sp[c * 50 + g * 25 + cc] * cvv[c];
        float v = (m[((size_t)b * 66 + g * 25 + cc) * HWC + pix] + s) * BNF_C;
        vals[cc] = v; mx = fmaxf(mx, v);
    }
    float den = 0.f;
    for (int cc = 0; cc < 25; ++cc) { vals[cc] = __expf(vals[cc] - mx); den += vals[cc]; }
    float d = 0.f;
    for (int cc = 0; cc < 25; ++cc)
        d += off[((size_t)b * 50 + g * 25 + cc) * HWC + pix] * vals[cc];
    disp[((size_t)b * 2 + g) * HWC + pix] = d / den;
}

// cvec -> local_context into fcon ch128..143 (f16, Ct=416); alpha -> d_out
__global__ void k_cvec(const float* __restrict__ m, h16* __restrict__ fcon,
                       float* __restrict__ alpha_out) {
    int i = blockIdx.x * blockDim.x + threadIdx.x;
    int tot = NBATCH * 16 * HWC;
    if (i >= tot) return;
    int pix = i % HWC; int c = (i / HWC) % 16; int b = i / (HWC * 16);
    float v = m[((size_t)b * 66 + 50 + c) * HWC + pix];
    fcon[hidx(b, 416, 128 + c, pix)] = (h16)(tanhf(v * 0.1f) * 10.f);
    if (c == 0) alpha_out[(size_t)b * HWC + pix] = sigm(v);
}

__global__ void k_final(const float* __restrict__ flow, const float* __restrict__ disp,
                        const float* __restrict__ alpha, float* __restrict__ flow_new) {
    int i = blockIdx.x * blockDim.x + threadIdx.x;
    int tot = NBATCH * 2 * HWC;
    if (i >= tot) return;
    int pix = i % HWC; int g = (i / HWC) % 2; int b = i / (HWC * 2);
    float f = flow[((size_t)b * 2 + g) * HWC + pix];
    float d = disp[((size_t)b * 2 + g) * HWC + pix] * (1.f / 64.f);
    float a = alpha[(size_t)b * HWC + pix];
    flow_new[((size_t)b * 2 + g) * HWC + pix] = (a > 0.1f) ? (f + d) : f;
}

// ---------------------------------------------------------------------------
extern "C" void kernel_launch(void* const* d_in, const int* in_sizes, int n_in,
                              void* d_out, int out_size, void* d_ws, size_t ws_size,
                              hipStream_t stream) {
    (void)in_sizes; (void)n_in; (void)out_size; (void)ws_size;
    const float* in_flow = (const float*)d_in[0];
    const float* in_cert = (const float*)d_in[1];
    const float* in_f1   = (const float*)d_in[2];
    const float* in_ctx  = (const float*)d_in[3];
    const float* in_f2   = (const float*)d_in[4];
    const float* w_score = (const float*)d_in[13];
    const float* sd_dw1  = (const float*)d_in[14];
    const float* sd_dw2  = (const float*)d_in[18];
    const float* skipW   = (const float*)d_in[32];
    const float* od2b    = (const float*)d_in[38];
    float* out = (float*)d_out;

    const size_t FLOW_OFF = 0, ALPHA_OFF = 16384, CERT_OFF = 24576, CN2_OFF = 40960;

    size_t cur = 0;
    char* base = (char*)d_ws;
    auto allocB = [&](size_t bytes) -> void* {
        cur = (cur + 255) & ~(size_t)255;
        void* p = base + cur; cur += bytes; return p;
    };
    auto allocF = [&](size_t n) { return (float*)allocB(n * 4); };
    auto allocH = [&](size_t n) { return (h16*)allocB(n * 2); };
    auto EL = [](size_t n) { return (int)((n + 255) / 256); };

    auto cvtW = [&](int idx, int O, int C, int Op, int Cp) -> const h16* {
        h16* p = allocH((size_t)Op * Cp);
        k_cvtW<<<EL((size_t)Op * Cp), 256, 0, stream>>>((const float*)d_in[idx], p, O, C, Op, Cp);
        return p;
    };

    const h16* w_ccf1  = cvtW(5, 128, 384, 128, 384);
    const h16* w_cof01 = cvtW(6, 160, 256, 160, 256);
    const h16* w_cof02 = cvtW(7, 160, 256, 160, 256);
    const h16* cf12_0  = cvtW(8, 256, 321, 256, 352);
    const h16* cf12_1  = cvtW(9, 256, 256, 256, 256);
    const h16* cf12_2  = cvtW(10, 256, 256, 256, 256);
    const h16* w_fcf   = cvtW(11, 256, 384, 256, 384);
    const h16* w_cof1cv= cvtW(12, 128, 256, 128, 256);
    const h16* sd_pw1  = cvtW(15, 32, 96, 32, 96);
    const h16* sd_m1a  = cvtW(16, 16, 32, 16, 32);
    const h16* sd_m1b  = cvtW(17, 16, 32, 16, 32);
    const h16* sd_pw2  = cvtW(19, 8, 16, 16, 32);
    const h16* sd_m2a  = cvtW(20, 2, 8, 16, 32);
    const h16* sd_m2b  = cvtW(21, 2, 8, 16, 32);
    const h16* w_cof2cv= cvtW(22, 131, 256, 144, 256);
    const h16* ce0a    = cvtW(23, 128, 166, 128, 192);
    const h16* ce0b    = cvtW(24, 128, 166, 128, 192);
    const h16* ce1     = cvtW(25, 128, 128, 128, 128);
    const h16* cvm0a   = cvtW(26, 66, 153, 80, 160);
    const h16* cvm0b   = cvtW(27, 66, 153, 80, 160);
    const h16* cvm1a   = cvtW(28, 66, 66, 80, 96);
    const h16* cvm1b   = cvtW(29, 66, 66, 80, 96);
    const h16* cvm2a   = cvtW(30, 66, 66, 80, 96);
    const h16* cvm2b   = cvtW(31, 66, 66, 80, 96);
    const h16* od0a    = cvtW(33, 50, 128, 64, 128);
    const h16* od0b    = cvtW(34, 50, 128, 64, 128);
    const h16* od1a    = cvtW(35, 50, 50, 64, 64);
    const h16* od1b    = cvtW(36, 50, 50, 64, 64);
    const h16* od2w    = cvtW(37, 50, 50, 64, 64);
    const h16* w_fcon  = cvtW(39, 144, 400, 144, 416);

    const size_t PB = (size_t)NBATCH * HWC;
    // f16 activation buffers, pair-interleaved layout, channel pad -> mult 32
    h16* feat2_h = allocH(PB * 256);
    h16* catCF   = allocH(PB * 384);   // [context | feat1]
    h16* catXF   = allocH(PB * 384);   // [ctx | feat_fused]
    h16* ffcat   = allocH(PB * 352);   // [f1c | f2s | overlap | pad]
    h16* X1h     = allocH(PB * 256);
    h16* X2h     = allocH(PB * 256);
    h16* Xsh     = allocH(PB * 96);
    h16* dw1h    = allocH(PB * 96);
    h16* pw1h    = allocH(PB * 32);
    h16* glu1h   = allocH(PB * 16);
    h16* dw2h    = allocH(PB * 32);    // 16 + pad
    h16* pw2h    = allocH(PB * 32);    // 8 + pad
    h16* zcat    = allocH(PB * 192);   // 166 + pad
    h16* zgh     = allocH(PB * 128);
    h16* mcat    = allocH(PB * 160);   // 153 + pad
    h16* m1h     = allocH(PB * 96);    // 66 + pad
    h16* m2h     = allocH(PB * 96);
    h16* o1h     = allocH(PB * 64);    // 50 + pad
    h16* o2h     = allocH(PB * 64);
    h16* fcon    = allocH(PB * 416);   // 400 + pad
    h16* rsh     = allocH(PB * 9 * 131);
    // f32 buffers
    float* ov_f   = allocF(PB);
    float* f2c_f  = allocF(PB * 160);
    float* ffu_f  = allocF(PB * 256);
    float* ctxn_f = allocF(PB * 256);
    float* f0_f   = allocF(PB * 128);
    float* ctxc_f = allocF(PB * 74);
    float* scr_f  = allocF(PB * 18);
    float* gA_f   = allocF(PB * 128);
    float* gB_f   = allocF(PB * 128);
    float* scale_f= allocF(PB * 2);
    float* f1cv_f = allocF(PB * 131);
    float* cv_f   = allocF(PB * 25);
    float* SV_f   = allocF(PB * 36);
    float* m_f    = allocF(PB * 66);
    float* off_f  = allocF(PB * 50);
    float* sp_f   = allocF(25 * 50);
    float* disp_f = allocF(PB * 2);

    auto gemm = [&](const h16* Wm, const h16* Xm, int XCt, int XCo,
                    float* Yf, int YfCt, int YfCo,
                    h16* Yh, int YhCt, int YhCo,
                    const float* addp, int addCt, int addCo,
                    const float* bias, int O, int Cp, int relu, float alpha) {
        dim3 g(HWC / 16, (O + 63) / 64, NBATCH);
        k_gemm<<<g, 128, 0, stream>>>(Wm, Xm, XCt, XCo, Yf, YfCt, YfCo,
                                      Yh, YhCt, YhCo, addp, addCt, addCo,
                                      bias, O, Cp, relu, alpha);
    };

    // ---- zero pad channels of concat buffers ----
    k_zero_h<<<EL(PB * 31), 256, 0, stream>>>(ffcat, 352, 321, 31);
    k_zero_h<<<EL(PB * 16), 256, 0, stream>>>(dw2h, 32, 16, 16);
    k_zero_h<<<EL(PB * 24), 256, 0, stream>>>(pw2h, 32, 8, 24);
    k_zero_h<<<EL(PB * 26), 256, 0, stream>>>(zcat, 192, 166, 26);
    k_zero_h<<<EL(PB * 7), 256, 0, stream>>>(mcat, 160, 153, 7);
    k_zero_h<<<EL(PB * 30), 256, 0, stream>>>(m1h, 96, 66, 30);
    k_zero_h<<<EL(PB * 30), 256, 0, stream>>>(m2h, 96, 66, 30);
    k_zero_h<<<EL(PB * 14), 256, 0, stream>>>(o1h, 64, 50, 14);
    k_zero_h<<<EL(PB * 14), 256, 0, stream>>>(o2h, 64, 50, 14);
    k_zero_h<<<EL(PB * 16), 256, 0, stream>>>(fcon, 416, 400, 16);

    // ---- input placement ----
    k_place_h<<<EL(PB * 128), 256, 0, stream>>>(in_ctx, 128, 0, catCF, 384, 0, 128);
    k_place_h<<<EL(PB * 256), 256, 0, stream>>>(in_f1, 256, 0, catCF, 384, 128, 256);
    k_place_h<<<EL(PB * 256), 256, 0, stream>>>(in_f2, 256, 0, feat2_h, 256, 0, 256);
    k_place_h<<<EL(PB * 128), 256, 0, stream>>>(in_ctx, 128, 0, fcon, 416, 0, 128);
    k_overlap<<<EL(PB), 256, 0, stream>>>(in_cert, ov_f, ffcat);

    // ---- fusion trunk ----
    gemm(w_ccf1, catCF, 384, 0, nullptr, 0, 0, catXF, 384, 0, nullptr, 0, 0, nullptr, 128, 384, 0, 1.f);
    gemm(w_cof01, feat2_h, 256, 0, f2c_f, 160, 0, nullptr, 0, 0, nullptr, 0, 0, nullptr, 160, 256, 1, BNF_C);
    k_gsample_flow<<<EL(PB * 160), 256, 0, stream>>>(f2c_f, in_flow, ffcat);
    gemm(w_cof02, catCF, 384, 128, nullptr, 0, 0, ffcat, 352, 0, nullptr, 0, 0, nullptr, 160, 256, 1, BNF_C);
    gemm(cf12_0, ffcat, 352, 0, nullptr, 0, 0, X1h, 256, 0, nullptr, 0, 0, nullptr, 256, 352, 1, BNF_C);
    gemm(cf12_1, X1h, 256, 0, nullptr, 0, 0, X2h, 256, 0, nullptr, 0, 0, nullptr, 256, 256, 1, BNF_C);
    gemm(cf12_2, X2h, 256, 0, ffu_f, 256, 0, nullptr, 0, 0, nullptr, 0, 0, nullptr, 256, 256, 0, BNF_C);
    k_place_h<<<EL(PB * 256), 256, 0, stream>>>(ffu_f, 256, 0, catXF, 384, 128, 256);
    k_place_h<<<EL(PB * 256), 256, 0, stream>>>(ffu_f, 256, 0, fcon, 416, 144, 256);
    gemm(w_fcf, catXF, 384, 0, ctxn_f, 256, 0, nullptr, 0, 0, nullptr, 0, 0, nullptr, 256, 384, 0, BNF_C);
    gemm(w_cof1cv, catCF, 384, 128, f0_f, 128, 0, nullptr, 0, 0, ctxn_f, 256, 0, nullptr, 128, 256, 0, 1.f);

    // ---- scale estimator ----
    k_build_ctxc<<<EL(PB * 74), 256, 0, stream>>>(ctxn_f, ov_f, ctxc_f);
    k_conv3x3<<<EL(PB * 18), 256, 0, stream>>>(ctxc_f, w_score, scr_f, 74, 18, BNF_C);
    k_score_sm_sc<<<EL(PB * 2), 256, 0, stream>>>(scr_f, in_flow, Xsh);
    k_place_h<<<EL(PB * 74), 256, 0, stream>>>(ctxc_f, 74, 0, Xsh, 96, 0, 74);
    k_ratio<<<EL(PB * 4), 256, 0, stream>>>(Xsh);
    k_dwconv<<<EL(PB * 96), 256, 0, stream>>>(Xsh, sd_dw1, dw1h, 96, 96, 96);
    gemm(sd_pw1, dw1h, 96, 0, nullptr, 0, 0, pw1h, 32, 0, nullptr, 0, 0, nullptr, 32, 96, 0, BNF_C);
    gemm(sd_m1a, pw1h, 32, 0, gA_f, 16, 0, nullptr, 0, 0, nullptr, 0, 0, nullptr, 16, 32, 0, 1.f);
    gemm(sd_m1b, pw1h, 32, 0, gB_f, 16, 0, nullptr, 0, 0, nullptr, 0, 0, nullptr, 16, 32, 0, 1.f);
    k_glu<<<EL(PB * 16), 256, 0, stream>>>(gA_f, gB_f, nullptr, 0, 0, glu1h, 16, 0, 16, 0);
    k_dwconv<<<EL(PB * 16), 256, 0, stream>>>(glu1h, sd_dw2, dw2h, 16, 16, 32);
    gemm(sd_pw2, dw2h, 32, 0, nullptr, 0, 0, pw2h, 32, 0, nullptr, 0, 0, nullptr, 8, 32, 0, BNF_C);
    gemm(sd_m2a, pw2h, 32, 0, gA_f, 2, 0, nullptr, 0, 0, nullptr, 0, 0, nullptr, 2, 32, 0, 1.f);
    gemm(sd_m2b, pw2h, 32, 0, gB_f, 2, 0, nullptr, 0, 0, nullptr, 0, 0, nullptr, 2, 32, 0, 1.f);
    k_glu<<<EL(PB * 2), 256, 0, stream>>>(gA_f, gB_f, scale_f, 2, 0, zcat, 192, 128, 2, 2);

    // ---- local correlation ----
    gemm(w_cof2cv, feat2_h, 256, 0, f1cv_f, 131, 0, nullptr, 0, 0, nullptr, 0, 0, nullptr, 131, 256, 0, 1.f);
    k_corr1<<<EL(PB * 25), 256, 0, stream>>>(f1cv_f, f0_f, in_flow, scale_f, cv_f, rsh);
    k_corr2<<<EL(PB * 9), 256, 0, stream>>>(rsh, SV_f);

    // ---- cv decoder ----
    k_place_h<<<EL(PB * 128), 256, 0, stream>>>(ctxn_f, 256, 128, zcat, 192, 0, 128);
    k_place_h<<<EL(PB * 36), 256, 0, stream>>>(SV_f, 36, 0, zcat, 192, 130, 36);
    gemm(ce0a, zcat, 192, 0, gA_f, 128, 0, nullptr, 0, 0, nullptr, 0, 0, nullptr, 128, 192, 0, 1.f);
    gemm(ce0b, zcat, 192, 0, gB_f, 128, 0, nullptr, 0, 0, nullptr, 0, 0, nullptr, 128, 192, 0, 1.f);
    k_glu<<<EL(PB * 128), 256, 0, stream>>>(gA_f, gB_f, nullptr, 0, 0, zgh, 128, 0, 128, 0);
    gemm(ce1, zgh, 128, 0, nullptr, 0, 0, mcat, 160, 0, nullptr, 0, 0, nullptr, 128, 128, 0, BNF_C);
    k_place_h<<<EL(PB * 25), 256, 0, stream>>>(cv_f, 25, 0, mcat, 160, 128, 25);
    gemm(cvm0a, mcat, 160, 0, gA_f, 66, 0, nullptr, 0, 0, nullptr, 0, 0, nullptr, 66, 160, 0, 1.f);
    gemm(cvm0b, mcat, 160, 0, gB_f, 66, 0, nullptr, 0, 0, nullptr, 0, 0, nullptr, 66, 160, 0, 1.f);
    k_glu<<<EL(PB * 66), 256, 0, stream>>>(gA_f, gB_f, nullptr, 0, 0, m1h, 96, 0, 66, 1);
    gemm(cvm1a, m1h, 96, 0, gA_f, 66, 0, nullptr, 0, 0, nullptr, 0, 0, nullptr, 66, 96, 0, 1.f);
    gemm(cvm1b, m1h, 96, 0, gB_f, 66, 0, nullptr, 0, 0, nullptr, 0, 0, nullptr, 66, 96, 0, 1.f);
    k_glu<<<EL(PB * 66), 256, 0, stream>>>(gA_f, gB_f, nullptr, 0, 0, m2h, 96, 0, 66, 1);
    gemm(cvm2a, m2h, 96, 0, gA_f, 66, 0, nullptr, 0, 0, nullptr, 0, 0, nullptr, 66, 96, 0, 1.f);
    gemm(cvm2b, m2h, 96, 0, gB_f, 66, 0, nullptr, 0, 0, nullptr, 0, 0, nullptr, 66, 96, 0, 1.f);
    k_glu<<<EL(PB * 66), 256, 0, stream>>>(gA_f, gB_f, m_f, 66, 0, nullptr, 0, 0, 66, 0);
    k_softplus<<<EL(1250), 256, 0, stream>>>(skipW, sp_f, 1250);
    gemm(od0a, mcat, 160, 0, gA_f, 50, 0, nullptr, 0, 0, nullptr, 0, 0, nullptr, 50, 128, 0, 1.f);
    gemm(od0b, mcat, 160, 0, gB_f, 50, 0, nullptr, 0, 0, nullptr, 0, 0, nullptr, 50, 128, 0, 1.f);
    k_glu<<<EL(PB * 50), 256, 0, stream>>>(gA_f, gB_f, nullptr, 0, 0, o1h, 64, 0, 50, 0);
    gemm(od1a, o1h, 64, 0, gA_f, 50, 0, nullptr, 0, 0, nullptr, 0, 0, nullptr, 50, 64, 0, 1.f);
    gemm(od1b, o1h, 64, 0, gB_f, 50, 0, nullptr, 0, 0, nullptr, 0, 0, nullptr, 50, 64, 0, 1.f);
    k_glu<<<EL(PB * 50), 256, 0, stream>>>(gA_f, gB_f, nullptr, 0, 0, o2h, 64, 0, 50, 0);
    gemm(od2w, o2h, 64, 0, off_f, 50, 0, nullptr, 0, 0, nullptr, 0, 0, od2b, 50, 64, 0, 1.f);
    k_decoder_final<<<EL(PB * 2), 256, 0, stream>>>(m_f, cv_f, sp_f, off_f, disp_f);

    // ---- outputs ----
    k_cvec<<<EL(PB * 16), 256, 0, stream>>>(m_f, fcon, out + ALPHA_OFF);
    gemm(w_fcon, fcon, 416, 0, out + CN2_OFF, 144, 0, nullptr, 0, 0, nullptr, 0, 0, nullptr, 144, 416, 0, BNF_C);
    k_final<<<EL(PB * 2), 256, 0, stream>>>(in_flow, disp_f, out + ALPHA_OFF, out + FLOW_OFF);
    k_copy_f<<<EL(PB * 2), 256, 0, stream>>>(in_cert, out + CERT_OFF, (int)(PB * 2));
    (void)FLOW_OFF; (void)CERT_OFF;
}
